// ChildSumTreeLSTM_33792802685043
// MI455X (gfx1250) — matrix operand order repaired and checked by reading, not compiled, not run on hardware
//
#include <hip/hip_runtime.h>
#include <hip/hip_fp16.h>
#include <hip/hip_bf16.h>

// ---------------------------------------------------------------------------
// ChildSum TreeLSTM on MI455X (gfx1250, wave32, WMMA).
// One fused GEMM per tree level:  A = [x | h_c0 | h_c1] (M x 768, f16 on the
// fly), Bmat = column-permuted fused weights (768 x 1280, f16, L2-resident),
// f32 WMMA accumulation, per-lane LSTM epilogue (no second pass).
// ---------------------------------------------------------------------------

typedef _Float16 v16h __attribute__((ext_vector_type(16)));
typedef _Float16 v8h  __attribute__((ext_vector_type(8)));
typedef float    v8f  __attribute__((ext_vector_type(8)));
typedef float    v4f  __attribute__((ext_vector_type(4)));

#define NB      8          // batch
#define NLEAF   16384
#define NNODES  32767      // 2*NLEAF - 1
#define DH      256
#define KFULL   768        // 256 x + 256 h0 + 256 h1
#define NOUT    1280       // 5 sections * 256 channels, column-permuted

// workspace layout (bytes)
#define WB_BYTES   ((size_t)NOUT * KFULL * 2)                 // 1,966,080
#define H_BYTES    ((size_t)NB * NNODES * DH * 2)             // 134,213,632
#define C_BYTES    ((size_t)NB * NNODES * DH * 4)             // 268,427,264
// total ~404.6 MB

__device__ __forceinline__ float sigm(float v) {
    return 1.0f / (1.0f + __expf(-v));
}

// ---------------------------------------------------------------------------
// Build permuted fused weight matrix, stored as Bmat^T : Wb[col][k], f16.
// col = g*320 + sec*64 + (ch%64); ch = g*64 + (col%320)%64; sec in {i,o,u,f0,f1}
// row k: [0,256)=x part, [256,512)=child0 part, [512,768)=child1 part.
// ---------------------------------------------------------------------------
__global__ void prep_wbig(const float* __restrict__ W_iou,
                          const float* __restrict__ U_iou,
                          const float* __restrict__ W_f,
                          const float* __restrict__ U_f,
                          _Float16* __restrict__ Wb) {
    int t = blockIdx.x * 256 + threadIdx.x;
    if (t >= NOUT * KFULL) return;
    int col = t / KFULL;
    int k   = t - col * KFULL;
    int g   = col / 320;
    int r   = col - g * 320;
    int sec = r / 64;
    int ch  = g * 64 + (r & 63);
    float v = 0.0f;
    if (sec < 3) {                       // i / o / u  -> W_iou, U_iou (both children)
        int cc = sec * 256 + ch;
        if (k < 256)       v = W_iou[k * 768 + cc];
        else if (k < 512)  v = U_iou[(k - 256) * 768 + cc];
        else               v = U_iou[(k - 512) * 768 + cc];
    } else if (sec == 3) {               // f gate, child 0
        if (k < 256)       v = W_f[k * 256 + ch];
        else if (k < 512)  v = U_f[(k - 256) * 256 + ch];
    } else {                             // f gate, child 1
        if (k < 256)       v = W_f[k * 256 + ch];
        else if (k >= 512) v = U_f[(k - 512) * 256 + ch];
    }
    Wb[(size_t)col * KFULL + k] = (_Float16)v;
}

// ---------------------------------------------------------------------------
// Per-level fused GEMM + LSTM epilogue.
//   grid.x : tiles of 64 rows over M = NB*n   (row = b*n + j, node = n-1+j)
//   grid.y : 4 channel groups of 64 channels
//   block  : 256 threads = 8 waves, 2 (M) x 4 (N); wave = 32 rows x NSEC tiles
// ---------------------------------------------------------------------------
template <bool LEAF>
__launch_bounds__(256)
__global__ void tree_level(const float*    __restrict__ x,
                           const _Float16* __restrict__ Wb,
                           const float*    __restrict__ b_iou,
                           const float*    __restrict__ b_f,
                           _Float16*       __restrict__ hbuf,
                           float*          __restrict__ cbuf,
                           int n, int nlog) {
    constexpr int NSEC   = LEAF ? 3 : 5;
    constexpr int KSTEPS = LEAF ? 8 : 24;   // K = 256 (leaf) or 768

    __shared__ _Float16 As[64][40];          // 64-row A tile per K-step, padded

    const int tid   = threadIdx.x;
    const int lane  = tid & 31;
    const int w     = tid >> 5;
    const int wm    = w & 1;                 // wave row block (0..1)
    const int wn    = w >> 1;                // wave channel block (0..3)
    const int g     = blockIdx.y;            // channel group (0..3)
    const int m0    = blockIdx.x * 64;
    const int M     = NB * n;
    const int nbase = n - 1;
    const int nmask = n - 1;

    v8f acc[2][NSEC];
#pragma unroll
    for (int mi = 0; mi < 2; ++mi)
#pragma unroll
        for (int s = 0; s < NSEC; ++s)
#pragma unroll
            for (int e = 0; e < 8; ++e) acc[mi][s][e] = 0.0f;

    // staging assignment: thread -> (row, 8-element K chunk)
    const int  srow   = tid >> 2;
    const int  skk    = (tid & 3) * 8;
    const int  rg     = m0 + srow;
    const bool svalid = rg < M;
    const int  sb     = rg >> nlog;          // batch
    const int  sj     = rg & nmask;          // node within level
    const int  snode  = nbase + sj;

    // WMMA fragment K sub-chunk base (contiguous K per lane half, same order for A and B)
    const int k0A  = (lane >> 4) * 16;       // A: K 0..15 / 16..31 contiguous
    const int k0B  = (lane >> 4) * 16;       // B: K 0..15 / 16..31 contiguous
    const int nidx = lane & 15;

    for (int ks = 0; ks < KSTEPS; ++ks) {
        __syncthreads();
        // ---- stage A tile (64 x 32 f16) ----
        v8h st;
#pragma unroll
        for (int e = 0; e < 8; ++e) st[e] = (_Float16)0.0f;
        if (svalid) {
            if (LEAF || ks < 8) {            // x slice, convert f32 -> f16
                const float* src = x + ((size_t)sb * NNODES + snode) * DH
                                     + ks * 32 + skk;
                v4f f0 = *(const v4f*)(src);
                v4f f1 = *(const v4f*)(src + 4);
#pragma unroll
                for (int e = 0; e < 4; ++e) {
                    st[e]     = (_Float16)f0[e];
                    st[4 + e] = (_Float16)f1[e];
                }
            } else {                         // child h, already f16
                const int child = 2 * snode + 1 + ((ks >= 16) ? 1 : 0);
                const _Float16* src = hbuf + ((size_t)sb * NNODES + child) * DH
                                           + (ks & 7) * 32 + skk;
                st = *(const v8h*)src;
            }
        }
        *(v8h*)&As[srow][skk] = st;
        __syncthreads();

        // ---- A fragments from LDS ----
        v16h a[2];
#pragma unroll
        for (int mi = 0; mi < 2; ++mi) {
            const int row = wm * 32 + mi * 16 + nidx;
            v8h lo = *(const v8h*)&As[row][k0A];
            v8h hi = *(const v8h*)&As[row][k0A + 8];
            a[mi] = __builtin_shufflevector(lo, hi, 0, 1, 2, 3, 4, 5, 6, 7,
                                                    8, 9, 10, 11, 12, 13, 14, 15);
        }

        // ---- B fragments from global (L2-resident) + WMMA ----
#pragma unroll
        for (int sec = 0; sec < NSEC; ++sec) {
            const int col = g * 320 + sec * 64 + wn * 16 + nidx;
            const _Float16* bp = Wb + (size_t)col * KFULL + ks * 32 + k0B;
            v16h bf = *(const v16h*)bp;
            acc[0][sec] = __builtin_amdgcn_wmma_f32_16x16x32_f16(
                false, a[0], false, bf, (short)0, acc[0][sec], false, false);
            acc[1][sec] = __builtin_amdgcn_wmma_f32_16x16x32_f16(
                false, a[1], false, bf, (short)0, acc[1][sec], false, false);
        }
    }

    // ---- per-lane LSTM epilogue (i,o,u,f0,f1 live in matching lanes/VGPRs) ----
    const int   ch = g * 64 + wn * 16 + nidx;
    const float bi = b_iou[ch];
    const float bo = b_iou[256 + ch];
    const float bu = b_iou[512 + ch];
    const float bff = LEAF ? 0.0f : b_f[ch];

#pragma unroll
    for (int mi = 0; mi < 2; ++mi) {
#pragma unroll
        for (int r = 0; r < 8; ++r) {
            const int lrow = wm * 32 + mi * 16 + r + ((lane >> 4) * 8);
            const int rg2  = m0 + lrow;
            if (rg2 >= M) continue;
            const int b = rg2 >> nlog;
            const int j = rg2 & nmask;
            const int p = nbase + j;

            const float i_ = acc[mi][0][r] + bi;
            const float o_ = acc[mi][1][r] + bo;
            const float u_ = acc[mi][2][r] + bu;
            float c = sigm(i_) * tanhf(u_);
            if (!LEAF) {
                const float f0 = acc[mi][3][r] + bff;
                const float f1 = acc[mi][4][r] + bff;
                const size_t cb = (size_t)b * NNODES * DH;
                const float c0 = cbuf[cb + (size_t)(2 * p + 1) * DH + ch];
                const float c1 = cbuf[cb + (size_t)(2 * p + 2) * DH + ch];
                c += sigm(f0) * c0 + sigm(f1) * c1;
            }
            const float h = sigm(o_) * tanhf(c);
            const size_t off = ((size_t)b * NNODES + p) * DH + ch;
            cbuf[off] = c;
            hbuf[off] = (_Float16)h;
        }
    }
}

// ---------------------------------------------------------------------------
// out[b] = concat(h_root, c_root)  (B x 512 f32)
// ---------------------------------------------------------------------------
__global__ void finalize(const _Float16* __restrict__ hbuf,
                         const float*    __restrict__ cbuf,
                         float*          __restrict__ out) {
    int t = blockIdx.x * 256 + threadIdx.x;
    if (t >= NB * 512) return;
    int b = t >> 9;
    int q = t & 511;
    size_t base = (size_t)b * NNODES * DH;   // root node = 0
    out[t] = (q < 256) ? (float)hbuf[base + q] : cbuf[base + (q - 256)];
}

extern "C" void kernel_launch(void* const* d_in, const int* in_sizes, int n_in,
                              void* d_out, int out_size, void* d_ws, size_t ws_size,
                              hipStream_t stream) {
    const float* x     = (const float*)d_in[0];
    const float* W_iou = (const float*)d_in[1];
    const float* b_iou = (const float*)d_in[2];
    const float* U_iou = (const float*)d_in[3];
    const float* W_f   = (const float*)d_in[4];
    const float* b_f   = (const float*)d_in[5];
    const float* U_f   = (const float*)d_in[6];

    char* ws = (char*)d_ws;
    _Float16* Wb   = (_Float16*)ws;
    _Float16* hbuf = (_Float16*)(ws + WB_BYTES);
    float*    cbuf = (float*)(ws + WB_BYTES + H_BYTES);

    // 1) fused / permuted weights, f16 (2 MB, stays in L2)
    prep_wbig<<<(NOUT * KFULL + 255) / 256, 256, 0, stream>>>(W_iou, U_iou, W_f, U_f, Wb);

    // 2) leaf level: n = 16384, K = 256 (x only)
    {
        const int n = NLEAF, nlog = 14;
        dim3 grid((NB * n + 63) / 64, 4);
        tree_level<true><<<grid, 256, 0, stream>>>(x, Wb, b_iou, b_f, hbuf, cbuf, n, nlog);
    }

    // 3) 14 internal levels, n = 8192 .. 1, K = 768
    for (int nlog = 13; nlog >= 0; --nlog) {
        const int n = 1 << nlog;
        dim3 grid((NB * n + 63) / 64, 4);
        tree_level<false><<<grid, 256, 0, stream>>>(x, Wb, b_iou, b_f, hbuf, cbuf, n, nlog);
    }

    // 4) gather root h/c
    finalize<<<(NB * 512 + 255) / 256, 256, 0, stream>>>(hbuf, cbuf, (float*)d_out);
}
